// EdgeAttentionAggregator_36197984370758
// MI455X (gfx1250) — matrix operand order repaired and checked by reading, not compiled, not run on hardware
//
#include <hip/hip_runtime.h>

typedef unsigned int   u32;
typedef unsigned short u16;

typedef __attribute__((ext_vector_type(16))) __bf16 v16bf;
typedef __attribute__((ext_vector_type(8)))  float  v8f;

struct __align__(16) Q16 { u32 x, y, z, w; };
struct __align__(8)  Q8  { u32 x, y; };
union FragU { v16bf v; Q16 q[2]; };

#define NUM_NODES 50000
#define C_DIM 128
#define H_DIM 256

__device__ __forceinline__ u16 f2bf(float f) {
  u32 u = __float_as_uint(f);
  u += 0x7fffu + ((u >> 16) & 1u);      // round-to-nearest-even
  return (u16)(u >> 16);
}

// Packed B-fragment: 16 contiguous halfs per lane (two b128 loads).
__device__ __forceinline__ v16bf load_frag_g(const u16* __restrict__ p) {
  FragU u;
  u.q[0] = *reinterpret_cast<const Q16*>(p);
  u.q[1] = *reinterpret_cast<const Q16*>(p + 8);
  return u.v;
}

// A-fragment from LDS tile (row-major bf16). Lane l<16: K offs {0..7,16..23},
// lane>=16: {8..15,24..31}; caller passes base already including the 0/8 skew.
__device__ __forceinline__ v16bf load_fragA_lds(const u16* p) {
  FragU u;
  u.q[0] = *reinterpret_cast<const Q16*>(p);
  u.q[1] = *reinterpret_cast<const Q16*>(p + 16);
  return u.v;
}

__device__ __forceinline__ v8f wmma_bf16(v16bf a, v16bf b, v8f c) {
  return __builtin_amdgcn_wmma_f32_16x16x32_bf16(false, a, false, b,
                                                 (short)0, c, false, false);
}

__global__ void __launch_bounds__(256)
zero_kernel(float* __restrict__ p, int n4) {
  const int i = blockIdx.x * blockDim.x + threadIdx.x;
  if (i < n4) {
    float4 z; z.x = 0.f; z.y = 0.f; z.z = 0.f; z.w = 0.f;
    reinterpret_cast<float4*>(p)[i] = z;
  }
}

// Pack a row-major fp32 weight [K x nCols] into bf16 fragment-major order:
// fragment f = kt*nTilesN + nt; within fragment: lane-major, 16 halfs/lane,
// half j -> K = kt*32 + (lane<16 ? 0 : 8) + (j<8 ? j : 8+j), col = nt*16 + (lane&15).
__global__ void __launch_bounds__(256)
pack_w_kernel(const float* __restrict__ src, u16* __restrict__ dst,
              int nTilesN, int nCols, int total) {
  const int t = blockIdx.x * blockDim.x + threadIdx.x;
  if (t >= total) return;
  const int f    = t >> 9;
  const int r    = t & 511;
  const int lane = r >> 4;
  const int j    = r & 15;
  const int kt   = f / nTilesN;
  const int nt   = f - kt * nTilesN;
  const int K    = kt * 32 + ((lane < 16) ? 0 : 8) + ((j < 8) ? j : 8 + j);
  const int col  = nt * 16 + (lane & 15);
  dst[t] = f2bf(src[(size_t)K * nCols + col]);
}

// att = sigmoid(ef @ att_w + att_b); scatter-add att*ef onto agg[src].
__global__ void __launch_bounds__(256)
att_scatter_kernel(const float* __restrict__ ef, const int* __restrict__ idx,
                   const float* __restrict__ att_w, const float* __restrict__ att_b,
                   float* __restrict__ agg, int E) {
  const int lane = threadIdx.x & 31;
  const int wave = threadIdx.x >> 5;
  const float4 w = *reinterpret_cast<const float4*>(att_w + lane * 4);
  const float ab = att_b[0];
  const int eBase = (blockIdx.x * 8 + wave) * 4;
  #pragma unroll
  for (int i = 0; i < 4; ++i) {
    const int e = eBase + i;
    if (e >= E) return;
    const float4 x = *reinterpret_cast<const float4*>(ef + (size_t)e * C_DIM + lane * 4);
    float p = x.x * w.x + x.y * w.y + x.z * w.z + x.w * w.w;
    #pragma unroll
    for (int off = 16; off > 0; off >>= 1) p += __shfl_xor(p, off, 32);
    const float att = 1.0f / (1.0f + __expf(-(p + ab)));
    const int s = idx[e];
    float* dst = agg + (size_t)s * C_DIM + lane * 4;
    __hip_atomic_fetch_add(dst + 0, x.x * att, __ATOMIC_RELAXED, __HIP_MEMORY_SCOPE_AGENT);
    __hip_atomic_fetch_add(dst + 1, x.y * att, __ATOMIC_RELAXED, __HIP_MEMORY_SCOPE_AGENT);
    __hip_atomic_fetch_add(dst + 2, x.z * att, __ATOMIC_RELAXED, __HIP_MEMORY_SCOPE_AGENT);
    __hip_atomic_fetch_add(dst + 3, x.w * att, __ATOMIC_RELAXED, __HIP_MEMORY_SCOPE_AGENT);
  }
}

// Fused MLP: out = relu((ef + agg[src]) @ w1 + b1) @ w2 + b2.
// 32 edges (2 M-tiles) per wave; each B fragment feeds 2 WMMAs; B loads are
// software-pipelined (distance-1 rotation) so loads overlap matrix math.
__global__ void __launch_bounds__(128)
mlp_kernel(const float* __restrict__ ef, const int* __restrict__ idx,
           const float* __restrict__ agg,
           const u16* __restrict__ pw1, const float* __restrict__ b1,
           const u16* __restrict__ pw2, const float* __restrict__ b2,
           float* __restrict__ out, int E) {
  __shared__ u16 lds[4 * 8192];                 // 16 KB per wave, 64 KB total
  const int lane = threadIdx.x & 31;
  const int wave = threadIdx.x >> 5;
  u16* tile = &lds[wave * 8192];
  const int eBase = (blockIdx.x * 4 + wave) * 32;
  if (eBase + 32 > E) return;                   // wave-uniform, EXEC stays all-ones

  const int M      = lane & 15;
  const int N      = lane & 15;
  const int koff   = (lane >= 16) ? 8 : 0;
  const int rowoff = (lane >= 16) ? 8 : 0;

  // ---- stage x = ef[e] + agg[src[e]] as bf16 [32][128] in LDS ----
  #pragma unroll 4
  for (int m = 0; m < 32; ++m) {
    const int e = eBase + m;
    const int s = idx[e];
    const float4 a = *reinterpret_cast<const float4*>(ef  + (size_t)e * C_DIM + lane * 4);
    const float4 g = *reinterpret_cast<const float4*>(agg + (size_t)s * C_DIM + lane * 4);
    Q8 p;
    p.x = (u32)f2bf(a.x + g.x) | ((u32)f2bf(a.y + g.y) << 16);
    p.y = (u32)f2bf(a.z + g.z) | ((u32)f2bf(a.w + g.w) << 16);
    *reinterpret_cast<Q8*>(tile + m * 128 + lane * 4) = p;
  }

  // ---- A fragments: 2 M-tiles x 4 K-tiles (held across all of GEMM1) ----
  v16bf afr[2][4];
  #pragma unroll
  for (int mt = 0; mt < 2; ++mt)
    #pragma unroll
    for (int kt = 0; kt < 4; ++kt)
      afr[mt][kt] = load_fragA_lds(tile + (mt * 16 + M) * 128 + kt * 32 + koff);

  // ---- GEMM1: h[32][256] = relu(x @ w1 + b1), 4 quarters of 4 N-tiles ----
  #pragma unroll
  for (int q = 0; q < 4; ++q) {
    v8f acc[2][4];
    #pragma unroll
    for (int nt = 0; nt < 4; ++nt) {
      const float bv = b1[(q * 4 + nt) * 16 + N];
      v8f tmp = {bv, bv, bv, bv, bv, bv, bv, bv};
      acc[0][nt] = tmp;
      acc[1][nt] = tmp;
    }
    v16bf bcur = load_frag_g(pw1 + (size_t)(q * 4) * 512 + lane * 16);  // kt=0,nt=0
    #pragma unroll
    for (int i = 0; i < 16; ++i) {
      const int kt = i >> 2, nt = i & 3;
      v16bf bnext = bcur;
      if (i < 15) {
        const int i2   = i + 1;
        const int frag = ((i2 >> 2) * 16) + q * 4 + (i2 & 3);
        bnext = load_frag_g(pw1 + (size_t)frag * 512 + lane * 16);
      }
      acc[0][nt] = wmma_bf16(afr[0][kt], bcur, acc[0][nt]);
      acc[1][nt] = wmma_bf16(afr[1][kt], bcur, acc[1][nt]);
      bcur = bnext;
    }
    #pragma unroll
    for (int mt = 0; mt < 2; ++mt)
      #pragma unroll
      for (int nt = 0; nt < 4; ++nt) {
        const int col = (q * 4 + nt) * 16 + N;
        #pragma unroll
        for (int r = 0; r < 8; ++r)
          tile[(mt * 16 + r + rowoff) * 256 + col] = f2bf(fmaxf(acc[mt][nt][r], 0.0f));
      }
  }

  // ---- GEMM2: out[32][128] = h @ w2 + b2, 2 halves of 4 N-tiles ----
  #pragma unroll
  for (int half = 0; half < 2; ++half) {
    v8f acc2[2][4];
    #pragma unroll
    for (int nt = 0; nt < 4; ++nt) {
      const float bv = b2[(half * 4 + nt) * 16 + N];
      v8f tmp = {bv, bv, bv, bv, bv, bv, bv, bv};
      acc2[0][nt] = tmp;
      acc2[1][nt] = tmp;
    }
    v16bf a2c0 = load_fragA_lds(tile + M * 256 + koff);
    v16bf a2c1 = load_fragA_lds(tile + (16 + M) * 256 + koff);
    v16bf bcur = load_frag_g(pw2 + (size_t)(half * 4) * 512 + lane * 16);  // kt=0,nt=0
    #pragma unroll
    for (int i = 0; i < 32; ++i) {
      const int kt = i >> 2, nt = i & 3;
      v16bf a2n0 = a2c0, a2n1 = a2c1;
      if (nt == 3 && kt < 7) {                  // prefetch next K-tile's A pair
        a2n0 = load_fragA_lds(tile + M * 256 + (kt + 1) * 32 + koff);
        a2n1 = load_fragA_lds(tile + (16 + M) * 256 + (kt + 1) * 32 + koff);
      }
      v16bf bnext = bcur;
      if (i < 31) {                             // prefetch next B fragment
        const int i2   = i + 1;
        const int frag = ((i2 >> 2) * 8) + half * 4 + (i2 & 3);
        bnext = load_frag_g(pw2 + (size_t)frag * 512 + lane * 16);
      }
      acc2[0][nt] = wmma_bf16(a2c0, bcur, acc2[0][nt]);
      acc2[1][nt] = wmma_bf16(a2c1, bcur, acc2[1][nt]);
      a2c0 = a2n0; a2c1 = a2n1; bcur = bnext;
    }
    #pragma unroll
    for (int mt = 0; mt < 2; ++mt)
      #pragma unroll
      for (int nt = 0; nt < 4; ++nt) {
        #pragma unroll
        for (int r = 0; r < 8; ++r)
          out[(size_t)(eBase + mt * 16 + r + rowoff) * C_DIM +
              (half * 4 + nt) * 16 + N] = acc2[mt][nt][r];
      }
  }
}

extern "C" void kernel_launch(void* const* d_in, const int* in_sizes, int n_in,
                              void* d_out, int out_size, void* d_ws, size_t ws_size,
                              hipStream_t stream) {
  const float* ef   = (const float*)d_in[0];
  const int*   idx  = (const int*)  d_in[1];   // row 0 = src
  const float* attw = (const float*)d_in[3];
  const float* attb = (const float*)d_in[4];
  const float* w1   = (const float*)d_in[5];
  const float* b1   = (const float*)d_in[6];
  const float* w2   = (const float*)d_in[7];
  const float* b2   = (const float*)d_in[8];
  float* out = (float*)d_out;
  const int E = in_sizes[0] / C_DIM;

  float* agg = (float*)d_ws;
  const size_t aggBytes = (size_t)NUM_NODES * C_DIM * sizeof(float);
  u16* pw1 = (u16*)((char*)d_ws + aggBytes);
  u16* pw2 = pw1 + C_DIM * H_DIM;

  const int n4 = NUM_NODES * C_DIM / 4;
  zero_kernel<<<(n4 + 255) / 256, 256, 0, stream>>>(agg, n4);
  pack_w_kernel<<<(C_DIM * H_DIM + 255) / 256, 256, 0, stream>>>(
      w1, pw1, H_DIM / 16, H_DIM, C_DIM * H_DIM);
  pack_w_kernel<<<(H_DIM * C_DIM + 255) / 256, 256, 0, stream>>>(
      w2, pw2, C_DIM / 16, C_DIM, H_DIM * C_DIM);
  att_scatter_kernel<<<(E + 31) / 32, 256, 0, stream>>>(ef, idx, attw, attb, agg, E);
  mlp_kernel<<<(E + 127) / 128, 128, 0, stream>>>(ef, idx, agg, pw1, b1, pw2, b2, out, E);
}
